// DynamicValueIteration_31224412241999
// MI455X (gfx1250) — matrix operand order repaired and checked by reading, not compiled, not run on hardware
//
#include <hip/hip_runtime.h>
#include <hip/hip_bf16.h>
#include <math.h>

typedef _Float16 half_t;
typedef __attribute__((ext_vector_type(16))) _Float16 v16h;
typedef __attribute__((ext_vector_type(8)))  float    v8f;

#define ACTION_CHANNELS 8
#define NUM_PLANS 2
#define IK 54           // NUM_PLANS * 27 : softmax/contraction dim
#define OGROUPS 16      // ACTION_CHANNELS * NUM_PLANS
#define KPAD 64         // K padded for 2 x WMMA 16x16x32
#define T_ 32
#define H_ 32
#define W_ 32
#define VOX_PER_B (T_ * H_ * W_)   // 32768
#define NBATCH 2

union V16HU { v16h v; float4 f4[2]; };

// ---------------- prep 1: rv = values + rewards ----------------
__global__ void dvi_rv_kernel(const float* __restrict__ values,
                              const float* __restrict__ rewards,
                              float* __restrict__ rv, int n) {
  int i = blockIdx.x * blockDim.x + threadIdx.x;
  if (i < n) rv[i] = values[i] + rewards[i];
}

// ------- prep 2: pack weight [864][54] fp32 -> Wr[16*64][64] f16, zero pad -------
__global__ void dvi_pack_w_kernel(const float* __restrict__ w,
                                  half_t* __restrict__ wr) {
  int idx = blockIdx.x * blockDim.x + threadIdx.x;  // 0..65535
  int row = idx >> 6;        // 0..1023 = o*64 + j
  int k   = idx & 63;
  int o   = row >> 6;
  int j   = row & 63;
  float v = 0.f;
  if (j < IK && k < IK) v = w[(o * IK + j) * IK + k];
  wr[idx] = (half_t)v;
}

// ---------------- fused: WMMA logits + softmax + dot + action max ----------------
__global__ __launch_bounds__(256)
void dvi_fused_kernel(const float* __restrict__ rv,
                      const half_t* __restrict__ wr,
                      float* __restrict__ out) {
  __shared__ __align__(16) half_t Uall[8][16 * KPAD];   // 16 KB total

  const int lane = threadIdx.x & 31;
  const int wave = threadIdx.x >> 5;
  const int tile = blockIdx.x * 8 + wave;   // 0..4095 : 16 voxels each
  const int vbase = tile << 4;
  const int b  = vbase >> 15;               // batch
  const int r  = vbase & (VOX_PER_B - 1);
  const int t  = r >> 10;
  const int h  = (r >> 5) & 31;
  const int w0 = r & 31;                    // 0 or 16

  half_t* U = &Uall[wave][0];

  // Gather U[n][k] = padded neighbor values, f16. 1024 entries / 32 lanes.
  for (int i = 0; i < 32; ++i) {
    int e = (i << 5) + lane;
    int n = e >> 6;
    int k = e & 63;
    float val = 0.f;
    if (k < IK) {
      int ci = k / 27;
      int rr = k % 27;
      int dz = rr / 9, dy = (rr % 9) / 3, dx = rr % 3;
      int tz = t + dz - 1, hy = h + dy - 1, wx = w0 + n + dx - 1;
      if ((unsigned)tz < 32u && (unsigned)hy < 32u && (unsigned)wx < 32u)
        val = rv[(((b * NUM_PLANS + ci) * T_ + tz) * H_ + hy) * W_ + wx];
    }
    U[n * KPAD + k] = (half_t)val;
  }
  __syncthreads();

  const int  col    = lane & 15;        // voxel within tile (N index)
  const bool hiHalf = lane >= 16;
  const int  hb     = hiHalf ? 8 : 0;   // C/D row offset for upper lanes

  // B fragments (same for every o-group): lane holds column `col`,
  // K = s*32 + (hiHalf?16:0) .. +15 contiguous halves (same k order as A).
  V16HU bfrag[2];
#pragma unroll
  for (int s = 0; s < 2; ++s) {
    const float4* p =
        (const float4*)&U[col * KPAD + s * 32 + (hiHalf ? 16 : 0)];
    bfrag[s].f4[0] = p[0];
    bfrag[s].f4[1] = p[1];
  }

  // Per-lane copy of u at its C-layout row indices j = rt*16 + hb + v.
  float u_reg[32];
#pragma unroll
  for (int rt = 0; rt < 4; ++rt)
#pragma unroll
    for (int v = 0; v < 8; ++v)
      u_reg[rt * 8 + v] = (float)U[col * KPAD + rt * 16 + hb + v];

  float best0 = -INFINITY, best1 = -INFINITY;

  for (int o = 0; o < OGROUPS; ++o) {
    v8f acc[4] = {};   // 64 rows x 16 cols of logits

#pragma unroll
    for (int s = 0; s < 2; ++s) {
#pragma unroll
      for (int rt = 0; rt < 4; ++rt) {
        // A layout: lane row M = col; halves = K s*32+2*hb..+15 contiguous
        const half_t* rp =
            wr + (size_t)((o * 64 + rt * 16 + col) * KPAD + s * 32 + 2 * hb);
        V16HU af;
        af.f4[0] = *(const float4*)(rp);
        af.f4[1] = *(const float4*)(rp + 8);
        acc[rt] = __builtin_amdgcn_wmma_f32_16x16x32_f16(
            false, af.v, false, bfrag[s].v, (short)0, acc[rt], false, false);
      }
    }

    // masked softmax over the 54 valid rows, fused with dot(u)
    float m = -INFINITY;
#pragma unroll
    for (int rt = 0; rt < 4; ++rt)
#pragma unroll
      for (int v = 0; v < 8; ++v) {
        int j = rt * 16 + hb + v;
        if (j < IK) m = fmaxf(m, acc[rt][v]);
      }
    m = fmaxf(m, __shfl_xor(m, 16, 32));

    float sum = 0.f, dot = 0.f;
#pragma unroll
    for (int rt = 0; rt < 4; ++rt)
#pragma unroll
      for (int v = 0; v < 8; ++v) {
        int j = rt * 16 + hb + v;
        if (j < IK) {
          float e = __expf(acc[rt][v] - m);
          sum += e;
          dot += e * u_reg[rt * 8 + v];
        }
      }
    sum += __shfl_xor(sum, 16, 32);
    dot += __shfl_xor(dot, 16, 32);

    float q = dot / sum;
    if (o < ACTION_CHANNELS) best0 = fmaxf(best0, q);
    else                     best1 = fmaxf(best1, q);
  }

  if (!hiHalf) {
    int vi = t * 1024 + h * 32 + w0 + col;
    out[(size_t)(b * NUM_PLANS + 0) * VOX_PER_B + vi] = best0;
    out[(size_t)(b * NUM_PLANS + 1) * VOX_PER_B + vi] = best1;
  }
}

extern "C" void kernel_launch(void* const* d_in, const int* in_sizes, int n_in,
                              void* d_out, int out_size, void* d_ws, size_t ws_size,
                              hipStream_t stream) {
  const float* values  = (const float*)d_in[0];
  const float* rewards = (const float*)d_in[1];
  const float* weight  = (const float*)d_in[2];
  float* out = (float*)d_out;

  // workspace layout: rv fp32 (512KB) | Wr f16 (128KB)
  float*  rv = (float*)d_ws;
  half_t* wr = (half_t*)((char*)d_ws + (size_t)NBATCH * NUM_PLANS * VOX_PER_B * sizeof(float));

  const int n_rv = NBATCH * NUM_PLANS * VOX_PER_B;  // 131072
  dvi_rv_kernel<<<(n_rv + 255) / 256, 256, 0, stream>>>(values, rewards, rv, n_rv);

  dvi_pack_w_kernel<<<(OGROUPS * 64 * KPAD) / 256, 256, 0, stream>>>(weight, wr);

  // 65536 voxels / 16 per wave / 8 waves per block = 512 blocks
  dvi_fused_kernel<<<512, 256, 0, stream>>>(rv, wr, out);
}
